// LSTMVoltageModel_16741782520005
// MI455X (gfx1250) — matrix operand order repaired and checked by reading, not compiled, not run on hardware
//
#include <hip/hip_runtime.h>
#include <hip/hip_bf16.h>

typedef __attribute__((ext_vector_type(16))) _Float16 v16h;
typedef __attribute__((ext_vector_type(8)))  float    v8f;
typedef __attribute__((ext_vector_type(2)))  float    v2f;

#define HID   32
#define TLEN  1024
#define BATCH 4096

__device__ __forceinline__ float fast_exp(float x) {
  return __builtin_amdgcn_exp2f(x * 1.4426950408889634f);   // v_exp_f32
}
__device__ __forceinline__ float fast_sigmoid(float x) {
  return __builtin_amdgcn_rcpf(1.0f + fast_exp(-x));        // v_rcp_f32
}
__device__ __forceinline__ float fast_tanh(float x) {
  // tanh(x) = 1 - 2/(exp(2x)+1)
  return 1.0f - 2.0f * __builtin_amdgcn_rcpf(1.0f + fast_exp(2.0f * x));
}

// One wave (32 lanes) owns 16 batch rows. Per timestep:
//   gates^T (128x16) = W_hh (128x32) * h^T (32x16)  + [x*W_ih + bias]
// as 8 tiles of v_wmma_f32_16x16x32_f16 (h-dependent, issued FIRST so they
// shadow the x load) chained into 8 tiles of v_wmma_f32_16x16x4_f32
// (A = [W_ih | bias | 0 | 0], B = [x;1;0;0], C = f16-WMMA result).
//
// W_hh rows are permuted inside the A-fragments so that the D-fragment of the
// new h is bit-for-bit the B-fragment layout of the next step's h^T:
//   logical unit at (tile tp, M=m):  u = 16*(m>>3) + 8*(tp&1) + (m&7)
// => the recurrent state stays in registers, no cross-lane traffic per step.
// x is streamed as b128 chunks with global_prefetch_b8 running ahead so the
// demand load hits the near cache.
__global__ __launch_bounds__(32)
void lstm_wmma_kernel(const float* __restrict__ x,
                      const float* __restrict__ W_ih,
                      const float* __restrict__ W_hh,
                      const float* __restrict__ b_ih,
                      const float* __restrict__ b_hh,
                      const float* __restrict__ W1, const float* __restrict__ b1,
                      const float* __restrict__ W2, const float* __restrict__ b2,
                      const float* __restrict__ W3, const float* __restrict__ b3,
                      float* __restrict__ out)
{
  const int lane = threadIdx.x & 31;
  const int lp   = lane & 15;     // A/B/C row-or-col index within 16
  const int half = lane >> 4;     // lane half selects K/M halves per ISA layouts
  const int b0   = blockIdx.x * 16;

  // ---- loop-invariant fragments ------------------------------------------
  v16h aW[8];   // W_hh tiles, f16 A-fragments (16x32 each)
  v2f  aX[8];   // [W_ih | bias] tiles, f32 A-fragments (16x4 each)
  #pragma unroll
  for (int tp = 0; tp < 8; ++tp) {
    // permuted W_hh row feeding tile tp at A-row M=lp
    const int r = 32 * (tp >> 1) + 16 * (lp >> 3) + 8 * (tp & 1) + (lp & 7);
    const float* row = W_hh + r * HID;
    #pragma unroll
    for (int j = 0; j < 16; ++j) {
      // f16 A 16x32, contiguous k order shared with hf: element j -> K = j + 16*half
      const int K = j + 16 * half;
      aW[tp][j] = (_Float16)row[K];
    }
    if (half == 0) {            // low lanes hold K=0 (W_ih) and K=1 (bias)
      aX[tp].x = W_ih[r];
      aX[tp].y = b_ih[r] + b_hh[r];
    } else {                    // high lanes hold K=2,3 -> zero padding
      aX[tp].x = 0.0f;
      aX[tp].y = 0.0f;
    }
  }

  // ---- recurrent state (registers only) ----------------------------------
  float c[2][8];                // cell state: unit u = 16*half + 8*j2 + v
  v16h  hf;                     // h^T B-fragment (f16), element j -> K = 16*half + j
  #pragma unroll
  for (int j2 = 0; j2 < 2; ++j2)
    #pragma unroll
    for (int v = 0; v < 8; ++v) c[j2][v] = 0.0f;
  #pragma unroll
  for (int j = 0; j < 16; ++j) hf[j] = (_Float16)0.0f;

  const float* xrow = x + (size_t)(b0 + lp) * TLEN;   // x is (B,1,T) row-major
  const float yb = half ? 0.0f : 1.0f;                // K=1 row of injection B

  for (int t = 0; t < TLEN; t += 4) {
    // run the prefetcher a few cachelines ahead (wrap keeps it in-bounds);
    // lowers to global_prefetch_b8: no counter, warms WGP$/L2 for the demand load
    __builtin_prefetch(xrow + ((t + 64) & (TLEN - 1)), 0, 3);

    const float4 xq = *(const float4*)(xrow + t);

    #pragma unroll
    for (int s = 0; s < 4; ++s) {
      const float xv = (s == 0) ? xq.x : (s == 1) ? xq.y : (s == 2) ? xq.z : xq.w;

      // ---- h-dependent WMMAs first: independent of the x load -------------
      v8f acc[8];
      #pragma unroll
      for (int tp = 0; tp < 8; ++tp) {
        v8f z = {};
        // acc = W_hh_tile * h^T   (f16 inputs, f32 accumulate)
        acc[tp] = __builtin_amdgcn_wmma_f32_16x16x32_f16(
            false, aW[tp], false, hf, (short)0, z, false, false);
      }

      // ---- x/bias injection second: acc += x*W_ih + bias (exact f32) ------
      v2f bX;
      bX.x = half ? 0.0f : xv;     // K=0 (low) / K=2 (high)
      bX.y = yb;                   // K=1 (low) / K=3 (high)
      #pragma unroll
      for (int tp = 0; tp < 8; ++tp) {
        acc[tp] = __builtin_amdgcn_wmma_f32_16x16x4_f32(
            false, aX[tp], false, bX, (short)0, acc[tp], false, false);
      }

      // activations + state update; new h lands directly in B-frag layout
      #pragma unroll
      for (int j2 = 0; j2 < 2; ++j2) {
        #pragma unroll
        for (int v = 0; v < 8; ++v) {
          const float iv = fast_sigmoid(acc[0 + j2][v]);
          const float fv = fast_sigmoid(acc[2 + j2][v]);
          const float gv = fast_tanh   (acc[4 + j2][v]);
          const float ov = fast_sigmoid(acc[6 + j2][v]);
          const float cc = fv * c[j2][v] + iv * gv;
          c[j2][v] = cc;
          const float hv = ov * fast_tanh(cc);
          hf[8 * j2 + v] = (_Float16)hv;   // B element j=8*j2+v <-> unit 16*half+j
        }
      }
    }
  }

  // ---- MLP head: 32 -> 16 -> 8 -> 1 on the final hidden state ------------
  __shared__ float hs[16 * HID];
  #pragma unroll
  for (int j = 0; j < 16; ++j) {
    const int u = 16 * half + j;           // hf element j holds unit 16*half+j
    hs[lp * HID + u] = (float)hf[j];
  }
  __syncthreads();

  if (half == 0) {                         // 16 lanes = 16 batch rows
    const float* hrow = hs + lp * HID;
    float a1[16];
    #pragma unroll
    for (int j = 0; j < 16; ++j) {
      float s = b1[j];
      #pragma unroll
      for (int k = 0; k < HID; ++k) s += W1[j * HID + k] * hrow[k];
      a1[j] = fmaxf(s, 0.0f);
    }
    float a2[8];
    #pragma unroll
    for (int j = 0; j < 8; ++j) {
      float s = b2[j];
      #pragma unroll
      for (int k = 0; k < 16; ++k) s += W2[j * 16 + k] * a1[k];
      a2[j] = fmaxf(s, 0.0f);
    }
    float s = b3[0];
    #pragma unroll
    for (int k = 0; k < 8; ++k) s += W3[k] * a2[k];
    out[b0 + lp] = s;
  }
}

extern "C" void kernel_launch(void* const* d_in, const int* in_sizes, int n_in,
                              void* d_out, int out_size, void* d_ws, size_t ws_size,
                              hipStream_t stream) {
  const float* x    = (const float*)d_in[0];
  const float* W_ih = (const float*)d_in[1];
  const float* W_hh = (const float*)d_in[2];
  const float* b_ih = (const float*)d_in[3];
  const float* b_hh = (const float*)d_in[4];
  const float* W1   = (const float*)d_in[5];
  const float* b1   = (const float*)d_in[6];
  const float* W2   = (const float*)d_in[7];
  const float* b2   = (const float*)d_in[8];
  const float* W3   = (const float*)d_in[9];
  const float* b3   = (const float*)d_in[10];
  float* out = (float*)d_out;

  dim3 grid(BATCH / 16);   // 256 waves: one wave per 16-sequence tile
  dim3 block(32);
  lstm_wmma_kernel<<<grid, block, 0, stream>>>(x, W_ih, W_hh, b_ih, b_hh,
                                               W1, b1, W2, b2, W3, b3, out);
}
